// SemanticPhaseBank_89618787598706
// MI455X (gfx1250) — matrix operand order repaired and checked by reading, not compiled, not run on hardware
//
#include <hip/hip_runtime.h>
#include <math.h>

// ---------------------------------------------------------------------------
// SemanticPhaseBank forward for MI455X (gfx1250), bf16 WMMA + f32 accumulate.
// B=4, S=2048, DIM=1024, N_CONCEPTS=2048, N_LAYERS=2
// GEMM: block tile 128x128, 8 waves, wave tile 32x64 (2x4 WMMA 16x16x32).
// ---------------------------------------------------------------------------

typedef __attribute__((ext_vector_type(16))) __bf16 v16bf;
typedef __attribute__((ext_vector_type(8)))  float  v8f;

#define ROWS   8192      // B*S
#define DIMF   1024
#define K2     2048      // 2*DIM (real||imag folded K)
#define NCON   2048

union QV { uint4 q[2]; v16bf v; };

__device__ __forceinline__ unsigned short f2bf(float f) {
    unsigned u = __float_as_uint(f);
    unsigned r = u + 0x7FFFu + ((u >> 16) & 1u);   // round-to-nearest-even
    return (unsigned short)(r >> 16);
}

// --- conversions ------------------------------------------------------------

// Xc[m, k] bf16 where k<1024 -> x[m,k,0], else x[m,k-1024,1]
__global__ void cvt_x_kernel(const float* __restrict__ x, unsigned short* __restrict__ Xc) {
    int i = blockIdx.x * blockDim.x + threadIdx.x;       // ROWS*K2 threads
    int m = i >> 11, k = i & 2047;
    int src = m * 2048 + ((k < DIMF) ? (k * 2) : ((k - DIMF) * 2 + 1));
    Xc[i] = f2bf(x[src]);
}

// Per concept n: Cc[n,k] (real||imag fold), CMt[j,n] (transpose of flat [n, d*2+p]),
// c_mag[n] = sqrt(sum(cr^2+ci^2) + 1e-8)
__global__ void cvt_concepts_kernel(const float* __restrict__ cm,
                                    unsigned short* __restrict__ Cc,
                                    unsigned short* __restrict__ CMt,
                                    float* __restrict__ c_mag) {
    __shared__ float red[256];
    int n = blockIdx.x, t = threadIdx.x;
    const float* row = cm + (size_t)n * 2048;
    float ss = 0.f;
    for (int j = t; j < 2048; j += 256) {
        float v = row[j];
        ss += v * v;
        CMt[(size_t)j * 2048 + n] = f2bf(v);
        int d = j >> 1, p = j & 1;
        int k = p ? (DIMF + d) : d;
        Cc[(size_t)n * 2048 + k] = f2bf(v);
    }
    red[t] = ss; __syncthreads();
    for (int s = 128; s > 0; s >>= 1) { if (t < s) red[t] += red[t + s]; __syncthreads(); }
    if (t == 0) c_mag[n] = sqrtf(red[0] + 1e-8f);
}

// Wc = bf16(W_layers[0], W_layers[1], W_out) row-major [o, d]
__global__ void cvt_w_kernel(const float* __restrict__ Wl, const float* __restrict__ Wo,
                             unsigned short* __restrict__ Wc) {
    int i = blockIdx.x * blockDim.x + threadIdx.x;       // 3*1024*1024 threads
    float v = (i < 2 * 1024 * 1024) ? Wl[i] : Wo[i - 2 * 1024 * 1024];
    Wc[i] = f2bf(v);
}

// --- softmax over concepts --------------------------------------------------

__global__ void softmax_kernel(const float* __restrict__ x, const float* __restrict__ scores,
                               const float* __restrict__ c_mag, unsigned short* __restrict__ attn) {
    __shared__ float red[256];
    int m = blockIdx.x, t = threadIdx.x;
    const float* xr = x + (size_t)m * 2048;
    float ss = 0.f;
#pragma unroll
    for (int j = 0; j < 8; j++) { float v = xr[t + j * 256]; ss += v * v; }
    red[t] = ss; __syncthreads();
    for (int s = 128; s > 0; s >>= 1) { if (t < s) red[t] += red[t + s]; __syncthreads(); }
    float xmag = sqrtf(red[0] + 1e-8f);
    __syncthreads();

    const float* srow = scores + (size_t)m * 2048;
    float c[8]; float mx = -1e30f;
#pragma unroll
    for (int j = 0; j < 8; j++) {
        int n = t + j * 256;
        float v = 2.0f * srow[n] / (xmag * c_mag[n] + 1e-8f);
        c[j] = v; mx = fmaxf(mx, v);
    }
    red[t] = mx; __syncthreads();
    for (int s = 128; s > 0; s >>= 1) { if (t < s) red[t] = fmaxf(red[t], red[t + s]); __syncthreads(); }
    mx = red[0]; __syncthreads();
    float sum = 0.f;
#pragma unroll
    for (int j = 0; j < 8; j++) { c[j] = __expf(c[j] - mx); sum += c[j]; }
    red[t] = sum; __syncthreads();
    for (int s = 128; s > 0; s >>= 1) { if (t < s) red[t] += red[t + s]; __syncthreads(); }
    float inv = 1.0f / red[0];
#pragma unroll
    for (int j = 0; j < 8; j++) attn[(size_t)m * 2048 + t + j * 256] = f2bf(c[j] * inv);
}

// --- per-component LayerNorm -> bf16 A-matrix [2*ROWS, 1024] ----------------

__global__ void layernorm_kernel(const float* __restrict__ Hin, const float* __restrict__ gamma,
                                 const float* __restrict__ beta, unsigned short* __restrict__ An) {
    __shared__ float red[256];
    int m = blockIdx.x, t = threadIdx.x;
    const float* row = Hin + (size_t)m * 2048;
    for (int p = 0; p < 2; p++) {
        float v[4]; float s = 0.f;
#pragma unroll
        for (int j = 0; j < 4; j++) { int d = t + j * 256; v[j] = row[d * 2 + p]; s += v[j]; }
        red[t] = s; __syncthreads();
        for (int st = 128; st > 0; st >>= 1) { if (t < st) red[t] += red[t + st]; __syncthreads(); }
        float mu = red[0] * (1.0f / 1024.0f); __syncthreads();
        float vs = 0.f;
#pragma unroll
        for (int j = 0; j < 4; j++) { float d0 = v[j] - mu; vs += d0 * d0; }
        red[t] = vs; __syncthreads();
        for (int st = 128; st > 0; st >>= 1) { if (t < st) red[t] += red[t + st]; __syncthreads(); }
        float rstd = rsqrtf(red[0] * (1.0f / 1024.0f) + 1e-5f); __syncthreads();
#pragma unroll
        for (int j = 0; j < 4; j++) {
            int d = t + j * 256;
            An[(size_t)(2 * m + p) * 1024 + d] = f2bf((v[j] - mu) * rstd * gamma[d] + beta[d]);
        }
    }
}

// --- WMMA GEMM: C[m,n] = sum_k A[m,k]*Bt[n,k], bf16 in / f32 acc ------------
// Block 128x128, 8 waves (4 wm x 2 wn), wave tile 32x64 (2x4 of 16x16x32).
// EPI 0: Out[m*N+n] = acc                             (scores)
// EPI 1: Out[i] = res[i] + 0.1*acc, i = m*N+n         (retrieval + residual)
// EPI 2: paired rows r=2m+p: bias, GELU(real) gate, residual -> Out[m*2048+n*2+p]
// EPI 3: paired rows: bias only -> Out[m*2048+n*2+p]  (final projection)

template <int EPI>
__global__ __launch_bounds__(256)
void gemm_kernel(const unsigned short* __restrict__ A, const unsigned short* __restrict__ Bt,
                 int M, int N, int K,
                 const float* __restrict__ bias, const float* __restrict__ res,
                 float* __restrict__ Out) {
    int w = threadIdx.x >> 5, lane = threadIdx.x & 31;
    int wm = w >> 1, wn = w & 1;
    int hi = lane >> 4, lr = lane & 15;
    int Mbase = blockIdx.y * 128 + wm * 32;
    int Nbase = blockIdx.x * 128 + wn * 64;

    v8f acc[2][4] = {};
    for (int kk = 0; kk < K; kk += 32) {
        QV a[2], b[4];
#pragma unroll
        for (int tm = 0; tm < 2; tm++) {
            const unsigned short* ap = A + (size_t)(Mbase + tm * 16 + lr) * K + kk + hi * 8;
            a[tm].q[0] = *(const uint4*)(ap);
            a[tm].q[1] = *(const uint4*)(ap + 16);
        }
#pragma unroll
        for (int tn = 0; tn < 4; tn++) {
            const unsigned short* bp = Bt + (size_t)(Nbase + tn * 16 + lr) * K + kk + hi * 8;
            b[tn].q[0] = *(const uint4*)(bp);
            b[tn].q[1] = *(const uint4*)(bp + 16);
        }
#pragma unroll
        for (int tm = 0; tm < 2; tm++)
#pragma unroll
            for (int tn = 0; tn < 4; tn++)
                acc[tm][tn] = __builtin_amdgcn_wmma_f32_16x16x32_bf16(
                    false, a[tm].v, false, b[tn].v, (short)0, acc[tm][tn], false, false);
    }

#pragma unroll
    for (int tm = 0; tm < 2; tm++)
#pragma unroll
        for (int tn = 0; tn < 4; tn++) {
            int n = Nbase + tn * 16 + lr;
            if (EPI == 0) {
#pragma unroll
                for (int v = 0; v < 8; v++) {
                    int mrow = Mbase + tm * 16 + hi * 8 + v;
                    Out[(size_t)mrow * N + n] = acc[tm][tn][v];
                }
            } else if (EPI == 1) {
#pragma unroll
                for (int v = 0; v < 8; v++) {
                    int mrow = Mbase + tm * 16 + hi * 8 + v;
                    size_t idx = (size_t)mrow * N + n;
                    Out[idx] = res[idx] + 0.1f * acc[tm][tn][v];
                }
            } else if (EPI == 2) {
#pragma unroll
                for (int v = 0; v < 8; v += 2) {
                    int r = Mbase + tm * 16 + hi * 8 + v;      // even row => p=0
                    int mrow = r >> 1;
                    float re = acc[tm][tn][v]     + bias[n];
                    float im = acc[tm][tn][v + 1] + bias[n];
                    float g = 0.5f * re * (1.0f + erff(re * 0.70710678118654752f));
                    size_t idx = (size_t)mrow * 2048 + n * 2;
                    Out[idx]     = res[idx]     + 0.1f * g * re;
                    Out[idx + 1] = res[idx + 1] + 0.1f * g * im;
                }
            } else {
#pragma unroll
                for (int v = 0; v < 8; v += 2) {
                    int r = Mbase + tm * 16 + hi * 8 + v;
                    int mrow = r >> 1;
                    float re = acc[tm][tn][v]     + bias[n];
                    float im = acc[tm][tn][v + 1] + bias[n];
                    size_t idx = (size_t)mrow * 2048 + n * 2;
                    Out[idx]     = re;
                    Out[idx + 1] = im;
                }
            }
        }
}

// ---------------------------------------------------------------------------

extern "C" void kernel_launch(void* const* d_in, const int* in_sizes, int n_in,
                              void* d_out, int out_size, void* d_ws, size_t ws_size,
                              hipStream_t stream) {
    (void)in_sizes; (void)n_in; (void)out_size; (void)ws_size;
    const float* x     = (const float*)d_in[0];
    const float* cm    = (const float*)d_in[1];
    const float* Wl    = (const float*)d_in[2];
    const float* bl    = (const float*)d_in[3];
    const float* gamma = (const float*)d_in[4];
    const float* beta  = (const float*)d_in[5];
    const float* Wo    = (const float*)d_in[6];
    const float* bo    = (const float*)d_in[7];
    const float* go    = (const float*)d_in[8];
    const float* bout  = (const float*)d_in[9];

    char* ws = (char*)d_ws;
    unsigned short* Xc   = (unsigned short*)(ws);                        // 32 MiB
    unsigned short* Cc   = (unsigned short*)(ws + (size_t)32  * 1048576);//  8 MiB
    unsigned short* CMt  = (unsigned short*)(ws + (size_t)40  * 1048576);//  8 MiB
    unsigned short* Wc   = (unsigned short*)(ws + (size_t)48  * 1048576);//  6 MiB
    float*          cmag = (float*)         (ws + (size_t)54  * 1048576);//  8 KiB
    float*          sc   = (float*)         (ws + (size_t)55  * 1048576);// 64 MiB
    unsigned short* attn = (unsigned short*)(ws + (size_t)119 * 1048576);// 32 MiB
    float*          H0   = (float*)         (ws + (size_t)151 * 1048576);// 64 MiB
    float*          H1   = (float*)         (ws + (size_t)215 * 1048576);// 64 MiB
    unsigned short* An   = (unsigned short*)(ws + (size_t)279 * 1048576);// 32 MiB -> 311 MiB total

    // 1) conversions / precompute
    cvt_x_kernel<<<(ROWS * K2) / 256, 256, 0, stream>>>(x, Xc);
    cvt_concepts_kernel<<<NCON, 256, 0, stream>>>(cm, Cc, CMt, cmag);
    cvt_w_kernel<<<(3 * 1024 * 1024) / 256, 256, 0, stream>>>(Wl, Wo, Wc);

    // 2) concept coherence scores: [8192,2048] x [2048,2048]^T
    dim3 gS(NCON / 128, ROWS / 128);
    gemm_kernel<0><<<gS, 256, 0, stream>>>(Xc, Cc, ROWS, NCON, K2, nullptr, nullptr, sc);

    // 3) softmax over concepts (with x_mag, c_mag normalization, scale 2.0)
    softmax_kernel<<<ROWS, 256, 0, stream>>>(x, sc, cmag, attn);

    // 4) retrieval + residual: H0 = x + 0.1 * attn @ concept_memory
    gemm_kernel<1><<<gS, 256, 0, stream>>>(attn, CMt, ROWS, K2, NCON, nullptr, x, H0);

    // 5) gated FFN layers
    float* Hc = H0; float* Hn = H1;
    dim3 gF(DIMF / 128, (2 * ROWS) / 128);
    for (int l = 0; l < 2; l++) {
        layernorm_kernel<<<ROWS, 256, 0, stream>>>(Hc, gamma + l * 1024, beta + l * 1024, An);
        gemm_kernel<2><<<gF, 256, 0, stream>>>(An, Wc + (size_t)l * 1024 * 1024,
                                               2 * ROWS, DIMF, DIMF, bl + l * 1024, Hc, Hn);
        float* t = Hc; Hc = Hn; Hn = t;
    }

    // 6) output LayerNorm + projection -> d_out (f32 [B,S,DIM,2])
    layernorm_kernel<<<ROWS, 256, 0, stream>>>(Hc, go, bout, An);
    gemm_kernel<3><<<gF, 256, 0, stream>>>(An, Wc + (size_t)2 * 1024 * 1024,
                                           2 * ROWS, DIMF, DIMF, bo, nullptr, (float*)d_out);
}
